// Block_61701500174566
// MI455X (gfx1250) — matrix operand order repaired and checked by reading, not compiled, not run on hardware
//
#include <hip/hip_runtime.h>
#include <hip/hip_bf16.h>
#include <math.h>

typedef _Float16 v16h  __attribute__((ext_vector_type(16)));
typedef _Float16 half8 __attribute__((ext_vector_type(8)));
typedef float    v8f   __attribute__((ext_vector_type(8)));

#define CCH 192
#define NHEAD 6
#define HDIM 32
#define NTOK 65536          // 2 * 32^3 tokens
#define TOK_PER_B 32768

// epilogue modes
constexpr int EPI_F16       = 0;  // out16 = acc + bias
constexpr int EPI_GELU_F16  = 1;  // out16 = gelu(acc + bias)
constexpr int EPI_RESID_TOK = 2;  // outf[token row] += acc + bias
constexpr int EPI_RESID_WIN = 3;  // outf[winrow->token] += acc + bias (inverse shift)
constexpr int EPI_CONV      = 4;  // relu((acc+bias)*invs*bn_g + bn_b) -> NCDHW scatter

// window-row -> token index (partition + roll composition; also its own inverse path)
__device__ __forceinline__ size_t winrow_to_token(int wr, int shifted) {
  int b = wr >> 15;
  int r = wr & 32767;
  int win = r >> 6, n = r & 63;
  int wh = win >> 6, ww = (win >> 3) & 7, wt = win & 7;
  int i = n >> 4, j = (n >> 2) & 3, k = n & 3;
  int h = wh * 4 + i, w = ww * 4 + j, t = wt * 4 + k;
  if (shifted) { h = (h + 2) & 31; w = (w + 2) & 31; t = (t + 2) & 31; }
  return ((size_t)b << 15) | (size_t)((h << 10) | (w << 5) | t);
}

// ------------- generic WMMA f16 GEMM: C[M,N] = A[M,K] * Bt[N,K]^T ------------------
// One wave computes a 32(M) x 64(N) tile: 2 A-fragments x 4 B-fragments
// -> 8 v_wmma_f32_16x16x32_f16 per 32-wide K step.
template <int EPI>
__global__ void gemm_wmma(const _Float16* __restrict__ A, int lda,
                          const _Float16* __restrict__ Bt, int ldb,
                          const float* __restrict__ bias,
                          int M, int N, int K,
                          _Float16* __restrict__ out16,
                          float* __restrict__ outf,
                          const float* __restrict__ aux0,
                          const float* __restrict__ aux1,
                          int flag) {
  const int lane = threadIdx.x & 31;
  const int wid  = (blockIdx.x * blockDim.x + threadIdx.x) >> 5;
  const int tilesN = N >> 6;
  const int totalTiles = (M >> 5) * tilesN;
  if (wid >= totalTiles) return;
  const int mt = wid / tilesN;       // 32-row tile index
  const int nt = wid % tilesN;       // 64-col tile index

  const int half16 = lane >> 4;
  const int l15 = lane & 15;
  const int koffA = half16 ? 8 : 0;
  const int koffB = half16 ? 8 : 0;

  const _Float16* Arow0 = A + (size_t)(mt * 32 + l15) * lda;
  const _Float16* Arow1 = Arow0 + (size_t)16 * lda;
  const _Float16* Bcol[4];
#pragma unroll
  for (int t = 0; t < 4; ++t)
    Bcol[t] = Bt + (size_t)(nt * 64 + t * 16 + l15) * ldb + koffB;

  v8f acc[2][4] = {};

  for (int kb = 0; kb < K; kb += 32) {
    __builtin_prefetch(Arow0 + kb + 128, 0, 1);
    __builtin_prefetch(Arow1 + kb + 128, 0, 1);
    half8 a0lo = *(const half8*)(Arow0 + kb + koffA);
    half8 a0hi = *(const half8*)(Arow0 + kb + koffA + 16);
    half8 a1lo = *(const half8*)(Arow1 + kb + koffA);
    half8 a1hi = *(const half8*)(Arow1 + kb + koffA + 16);
    v16h a0, a1;
#pragma unroll
    for (int i = 0; i < 8; ++i) {
      a0[i] = a0lo[i]; a0[i + 8] = a0hi[i];
      a1[i] = a1lo[i]; a1[i + 8] = a1hi[i];
    }
#pragma unroll
    for (int t = 0; t < 4; ++t) {
      half8 blo = *(const half8*)(Bcol[t] + kb);
      half8 bhi = *(const half8*)(Bcol[t] + kb + 16);
      v16h b;
#pragma unroll
      for (int i = 0; i < 8; ++i) { b[i] = blo[i]; b[i + 8] = bhi[i]; }
      acc[0][t] = __builtin_amdgcn_wmma_f32_16x16x32_f16(
          false, a0, false, b, (short)0, acc[0][t], false, false);
      acc[1][t] = __builtin_amdgcn_wmma_f32_16x16x32_f16(
          false, a1, false, b, (short)0, acc[1][t], false, false);
    }
  }

#pragma unroll
  for (int s = 0; s < 2; ++s) {
#pragma unroll
    for (int t = 0; t < 4; ++t) {
      const int col = nt * 64 + t * 16 + l15;
      const float bv = bias ? bias[col] : 0.0f;
#pragma unroll
      for (int r = 0; r < 8; ++r) {
        const int m = mt * 32 + s * 16 + r + 8 * half16;
        float v = acc[s][t][r] + bv;
        if (EPI == EPI_F16) {
          out16[(size_t)m * N + col] = (_Float16)v;
        } else if (EPI == EPI_GELU_F16) {
          v = 0.5f * v * (1.0f + erff(v * 0.70710678f));
          out16[(size_t)m * N + col] = (_Float16)v;
        } else if (EPI == EPI_RESID_TOK) {
          outf[(size_t)m * N + col] += v;
        } else if (EPI == EPI_RESID_WIN) {
          size_t tok = winrow_to_token(m, flag);
          outf[tok * CCH + col] += v;
        } else { // EPI_CONV: eval-mode BN + ReLU, scatter to (B, OUT, H, W, T)
          float y = v * 0.99999500f * aux0[col] + aux1[col];
          int b = m >> 15;
          int tokloc = m & 32767;
          outf[(((size_t)b * CCH + col) << 15) | (size_t)tokloc] = fmaxf(y, 0.0f);
        }
      }
    }
  }
}

// ---------------- LayerNorm over 192 channels, one wave per row --------------------
__global__ void ln_rows(const float* __restrict__ xf,
                        const float* __restrict__ g,
                        const float* __restrict__ bt,
                        _Float16* __restrict__ out,
                        int partitioned, int shifted) {
  const int lane = threadIdx.x & 31;
  const int wr = (blockIdx.x * blockDim.x + threadIdx.x) >> 5;
  if (wr >= NTOK) return;
  size_t src = partitioned ? winrow_to_token(wr, shifted) : (size_t)wr;
  const float* xp = xf + src * CCH;
  float v[6];
  float s = 0.0f;
#pragma unroll
  for (int j = 0; j < 6; ++j) { v[j] = xp[lane + 32 * j]; s += v[j]; }
#pragma unroll
  for (int o = 16; o > 0; o >>= 1) s += __shfl_xor(s, o, 32);
  const float mu = s * (1.0f / 192.0f);
  float q = 0.0f;
#pragma unroll
  for (int j = 0; j < 6; ++j) { float d = v[j] - mu; q += d * d; }
#pragma unroll
  for (int o = 16; o > 0; o >>= 1) q += __shfl_xor(q, o, 32);
  const float inv = rsqrtf(q * (1.0f / 192.0f) + 1e-5f);
  _Float16* op = out + (size_t)wr * CCH;
#pragma unroll
  for (int j = 0; j < 6; ++j) {
    int c = lane + 32 * j;
    op[c] = (_Float16)((v[j] - mu) * inv * g[c] + bt[c]);
  }
}

// ---------------- attention-side precomputed tables --------------------------------
__device__ __forceinline__ int region3(int x) { return x < 28 ? 0 : (x < 30 ? 1 : 2); }

// rel-pos bias expanded into WMMA C-fragment order:
// biasSw[((head*4 + wv)*32 + lane)*32 + nt*8 + r]
__global__ void expand_bias(const float* __restrict__ rpb, float* __restrict__ biasSw) {
  int idx = blockIdx.x * blockDim.x + threadIdx.x;
  if (idx >= NHEAD * 4096) return;
  int r   = idx & 7;
  int nt  = (idx >> 3) & 3;
  int lane = (idx >> 5) & 31;
  int wv  = (idx >> 10) & 3;
  int head = idx >> 12;
  int half16 = lane >> 4, l15 = lane & 15;
  int qn = wv * 16 + r + 8 * half16;
  int km = nt * 16 + l15;
  int i1 = qn >> 4, j1 = (qn >> 2) & 3, k1 = qn & 3;
  int i2 = km >> 4, j2 = (km >> 2) & 3, k2 = km & 3;
  int ridx = (i1 - i2 + 3) * 49 + (j1 - j2 + 3) * 7 + (k1 - k2 + 3);
  biasSw[idx] = rpb[ridx * NHEAD + head];
}

// shift-window region code per (window, token-in-window): codeTbl[w512*64 + n]
__global__ void build_codes(int* __restrict__ codeTbl) {
  int idx = blockIdx.x * blockDim.x + threadIdx.x;
  if (idx >= 512 * 64) return;
  int w512 = idx >> 6, n = idx & 63;
  int wh = ((w512 >> 6) & 7) * 4, ww = ((w512 >> 3) & 7) * 4, wt = (w512 & 7) * 4;
  codeTbl[idx] = region3(wh + (n >> 4)) * 9 + region3(ww + ((n >> 2) & 3)) * 3
               + region3(wt + (n & 3));
}

// ---------------- windowed attention: one block per (window, head) ------------------
__global__ void attn_wmma(const _Float16* __restrict__ qkv,
                          const float* __restrict__ biasSw,
                          const int* __restrict__ codeTbl,
                          _Float16* __restrict__ o16,
                          int shifted) {
  __shared__ _Float16 P[64][72];    // softmax probs (padded for 16B-aligned frags)
  __shared__ _Float16 Vt[32][72];   // V transposed: Vt[d][k]
  const int win  = blockIdx.x / NHEAD;
  const int head = blockIdx.x % NHEAD;
  const int lane = threadIdx.x & 31;
  const int wv   = threadIdx.x >> 5;   // 4 waves, 16 query rows each
  const int half16 = lane >> 4;
  const int l15 = lane & 15;
  const int mbase = wv * 16;
  const size_t winBase = (size_t)win * 64;
  const int w512 = win & 511;
  const int koffA = half16 ? 8 : 0;
  const int koffB = half16 ? 8 : 0;

  // ---- stage V transposed into LDS (coalesced global reads) ----
  const _Float16* Vbase = qkv + winBase * 576 + (2 * NHEAD + head) * HDIM;
  for (int idx = threadIdx.x; idx < 2048; idx += 128) {
    int k = idx >> 5, d = idx & 31;
    Vt[d][k] = Vbase[(size_t)k * 576 + d];
  }

  // ---- S = Q * K^T (per wave: 16 x 64, K = 32) ----
  const _Float16* Qrow = qkv + (winBase + mbase + l15) * 576 + head * HDIM;
  half8 qlo = *(const half8*)(Qrow + koffA);
  half8 qhi = *(const half8*)(Qrow + koffA + 16);
  v16h a;
#pragma unroll
  for (int i = 0; i < 8; ++i) { a[i] = qlo[i]; a[i + 8] = qhi[i]; }

  v8f s[4] = {};
#pragma unroll
  for (int nt = 0; nt < 4; ++nt) {
    const _Float16* Krow = qkv + (winBase + nt * 16 + l15) * 576
                               + (NHEAD + head) * HDIM + koffB;
    half8 blo = *(const half8*)Krow;
    half8 bhi = *(const half8*)(Krow + 16);
    v16h b;
#pragma unroll
    for (int i = 0; i < 8; ++i) { b[i] = blo[i]; b[i + 8] = bhi[i]; }
    s[nt] = __builtin_amdgcn_wmma_f32_16x16x32_f16(
        false, a, false, b, (short)0, s[nt], false, false);
  }

  // ---- scale + swizzled bias (contiguous 128B per lane) + branch-free mask ----
  const float scale = 0.1767766953f;   // 32^-0.5
  const float maskAmt = shifted ? 100.0f : 0.0f;
  const float* bp = biasSw + (size_t)(((head * 4 + wv) * 32 + lane) * 32);
  const int* cbase = codeTbl + w512 * 64;
  int kc[4], qc[8];
#pragma unroll
  for (int nt = 0; nt < 4; ++nt) kc[nt] = cbase[nt * 16 + l15];
#pragma unroll
  for (int r = 0; r < 8; ++r) qc[r] = cbase[mbase + r + 8 * half16];
#pragma unroll
  for (int nt = 0; nt < 4; ++nt) {
#pragma unroll
    for (int r = 0; r < 8; ++r) {
      float v = s[nt][r] * scale + bp[nt * 8 + r];
      v -= (qc[r] != kc[nt]) ? maskAmt : 0.0f;
      s[nt][r] = v;
    }
  }

  // ---- row softmax: a row lives in one lane-half (16 lanes), xor masks 1/2/4/8 ----
#pragma unroll
  for (int r = 0; r < 8; ++r) {
    float mx = s[0][r];
#pragma unroll
    for (int nt = 1; nt < 4; ++nt) mx = fmaxf(mx, s[nt][r]);
#pragma unroll
    for (int o = 8; o > 0; o >>= 1) mx = fmaxf(mx, __shfl_xor(mx, o, 32));
    float sum = 0.0f;
#pragma unroll
    for (int nt = 0; nt < 4; ++nt) {
      float e = __expf(s[nt][r] - mx);
      s[nt][r] = e;
      sum += e;
    }
#pragma unroll
    for (int o = 8; o > 0; o >>= 1) sum += __shfl_xor(sum, o, 32);
    float inv = 1.0f / sum;
#pragma unroll
    for (int nt = 0; nt < 4; ++nt) s[nt][r] *= inv;
  }

  // ---- stage P (f16) in LDS for A-fragment re-layout ----
#pragma unroll
  for (int nt = 0; nt < 4; ++nt)
#pragma unroll
    for (int r = 0; r < 8; ++r)
      P[mbase + r + 8 * half16][nt * 16 + l15] = (_Float16)s[nt][r];
  __syncthreads();   // covers both P and Vt staging

  // ---- O = P * V (per wave: 16 x 32, K = 64 -> two k-steps), V from LDS ----
  v8f oacc[2] = {};
  for (int kb = 0; kb < 64; kb += 32) {
    half8 plo = *(const half8*)(&P[mbase + l15][kb + koffA]);
    half8 phi = *(const half8*)(&P[mbase + l15][kb + koffA + 16]);
    v16h pa;
#pragma unroll
    for (int i = 0; i < 8; ++i) { pa[i] = plo[i]; pa[i + 8] = phi[i]; }
#pragma unroll
    for (int dt = 0; dt < 2; ++dt) {
      int dcol = dt * 16 + l15;
      half8 vlo = *(const half8*)(&Vt[dcol][kb + koffB]);
      half8 vhi = *(const half8*)(&Vt[dcol][kb + koffB + 16]);
      v16h b;
#pragma unroll
      for (int i = 0; i < 8; ++i) { b[i] = vlo[i]; b[i + 8] = vhi[i]; }
      oacc[dt] = __builtin_amdgcn_wmma_f32_16x16x32_f16(
          false, pa, false, b, (short)0, oacc[dt], false, false);
    }
  }
#pragma unroll
  for (int dt = 0; dt < 2; ++dt)
#pragma unroll
    for (int r = 0; r < 8; ++r) {
      int m = mbase + r + 8 * half16;
      o16[(winBase + m) * CCH + head * HDIM + dt * 16 + l15] = (_Float16)oacc[dt][r];
    }
}

// ---------------- small utility kernels --------------------------------------------
__global__ void transpose_in(const float* __restrict__ x, float* __restrict__ xf) {
  int idx = blockIdx.x * blockDim.x + threadIdx.x;
  if (idx >= 2 * CCH * TOK_PER_B) return;
  int b = idx / (CCH * TOK_PER_B);
  int r = idx - b * (CCH * TOK_PER_B);
  int c = r / TOK_PER_B;
  int tok = r - c * TOK_PER_B;
  xf[((size_t)b * TOK_PER_B + tok) * CCH + c] = x[idx];
}

__global__ void convert_wT(const float* __restrict__ w, _Float16* __restrict__ wt,
                           int K, int N) {           // w: K x N -> wt: N x K (f16)
  int idx = blockIdx.x * blockDim.x + threadIdx.x;
  if (idx >= K * N) return;
  int k = idx / N, n = idx - k * N;
  wt[(size_t)n * K + k] = (_Float16)w[idx];
}

__global__ void convert_w(const float* __restrict__ w, _Float16* __restrict__ o, int n) {
  int idx = blockIdx.x * blockDim.x + threadIdx.x;
  if (idx < n) o[idx] = (_Float16)w[idx];
}

__global__ void dwconv(const float* __restrict__ xf, const float* __restrict__ dww,
                       const float* __restrict__ dwb, _Float16* __restrict__ out) {
  int idx = blockIdx.x * blockDim.x + threadIdx.x;
  if (idx >= NTOK * CCH) return;
  int c = idx % CCH;
  int row = idx / CCH;
  int tok = row & 32767;
  int b = row >> 15;
  int h = tok >> 10, w = (tok >> 5) & 31, t = tok & 31;
  float acc = dwb[c];
  for (int dh = -1; dh <= 1; ++dh) {
    int hh = h + dh; if (hh < 0 || hh > 31) continue;
    for (int dw = -1; dw <= 1; ++dw) {
      int w2 = w + dw; if (w2 < 0 || w2 > 31) continue;
      for (int dt = -1; dt <= 1; ++dt) {
        int tt = t + dt; if (tt < 0 || tt > 31) continue;
        acc += xf[(((size_t)b << 15) + (size_t)((hh << 10) | (w2 << 5) | tt)) * CCH + c]
             * dww[c * 27 + (dh + 1) * 9 + (dw + 1) * 3 + (dt + 1)];
      }
    }
  }
  out[idx] = (_Float16)acc;
}

// ---------------- host driver ------------------------------------------------------
extern "C" void kernel_launch(void* const* d_in, const int* in_sizes, int n_in,
                              void* d_out, int out_size, void* d_ws, size_t ws_size,
                              hipStream_t stream) {
  (void)in_sizes; (void)n_in; (void)out_size; (void)ws_size;
  const float* x      = (const float*)d_in[0];
  const float* n1g    = (const float*)d_in[1];
  const float* n1b    = (const float*)d_in[2];
  const float* qkv_w  = (const float*)d_in[3];
  const float* qkv_b  = (const float*)d_in[4];
  const float* rpb    = (const float*)d_in[5];
  const float* proj_w = (const float*)d_in[6];
  const float* proj_b = (const float*)d_in[7];
  const float* n2g    = (const float*)d_in[8];
  const float* n2b    = (const float*)d_in[9];
  const float* fc1_w  = (const float*)d_in[10];
  const float* fc1_b  = (const float*)d_in[11];
  const float* fc2_w  = (const float*)d_in[12];
  const float* fc2_b  = (const float*)d_in[13];
  const float* dw_w   = (const float*)d_in[14];
  const float* dw_b   = (const float*)d_in[15];
  const float* pw_w   = (const float*)d_in[16];
  const float* pw_b   = (const float*)d_in[17];
  const float* bn_g   = (const float*)d_in[18];
  const float* bn_b   = (const float*)d_in[19];

  uint8_t* ws = (uint8_t*)d_ws;
  size_t off = 0;
  auto carve = [&](size_t bytes) -> void* {
    void* p = ws + off;
    off += (bytes + 255) & ~(size_t)255;
    return p;
  };
  float*     xf    = (float*)    carve((size_t)NTOK * CCH * 4);
  _Float16*  h16   = (_Float16*) carve((size_t)NTOK * CCH * 2);
  _Float16*  qkv16 = (_Float16*) carve((size_t)NTOK * 576 * 2);
  _Float16*  o16   = (_Float16*) carve((size_t)NTOK * CCH * 2);
  _Float16*  hid16 = (_Float16*) carve((size_t)NTOK * 768 * 2);
  float*     biasSw = (float*)   carve((size_t)NHEAD * 4096 * 4);
  int*       codeTbl = (int*)    carve((size_t)512 * 64 * 4);
  _Float16 *qkvT[4], *projT[4], *fc1T[4], *fc2T[4];
  for (int i = 0; i < 4; ++i) {
    qkvT[i] = (_Float16*)carve(192 * 576 * 2);
    projT[i] = (_Float16*)carve(192 * 192 * 2);
    fc1T[i] = (_Float16*)carve(192 * 768 * 2);
    fc2T[i] = (_Float16*)carve(768 * 192 * 2);
  }
  _Float16* pw16 = (_Float16*)carve(192 * 192 * 2);

  auto cvtT = [&](const float* w, _Float16* wt, int K, int N) {
    int n = K * N;
    convert_wT<<<(n + 255) / 256, 256, 0, stream>>>(w, wt, K, N);
  };
  for (int i = 0; i < 4; ++i) {
    cvtT(qkv_w  + (size_t)i * 192 * 576, qkvT[i], 192, 576);
    cvtT(proj_w + (size_t)i * 192 * 192, projT[i], 192, 192);
    cvtT(fc1_w  + (size_t)i * 192 * 768, fc1T[i], 192, 768);
    cvtT(fc2_w  + (size_t)i * 768 * 192, fc2T[i], 768, 192);
  }
  // pw_w is (OUT, C) == N x K already
  convert_w<<<(192 * 192 + 255) / 256, 256, 0, stream>>>(pw_w, pw16, 192 * 192);
  build_codes<<<(512 * 64 + 255) / 256, 256, 0, stream>>>(codeTbl);

  transpose_in<<<(2 * CCH * TOK_PER_B + 255) / 256, 256, 0, stream>>>(x, xf);

  auto gB = [](int M, int N) { return ((M / 32) * (N / 64) + 7) / 8; };

  for (int i = 0; i < 4; ++i) {
    const int shifted = i & 1;
    ln_rows<<<NTOK / 8, 256, 0, stream>>>(xf, n1g + i * 192, n1b + i * 192,
                                          h16, 1, shifted);
    gemm_wmma<EPI_F16><<<gB(NTOK, 576), 256, 0, stream>>>(
        h16, 192, qkvT[i], 192, qkv_b + i * 576, NTOK, 576, 192,
        qkv16, nullptr, nullptr, nullptr, 0);
    expand_bias<<<(NHEAD * 4096 + 255) / 256, 256, 0, stream>>>(
        rpb + (size_t)i * 343 * 6, biasSw);
    attn_wmma<<<1024 * NHEAD, 128, 0, stream>>>(qkv16, biasSw, codeTbl, o16, shifted);
    gemm_wmma<EPI_RESID_WIN><<<gB(NTOK, 192), 256, 0, stream>>>(
        o16, 192, projT[i], 192, proj_b + i * 192, NTOK, 192, 192,
        nullptr, xf, nullptr, nullptr, shifted);
    ln_rows<<<NTOK / 8, 256, 0, stream>>>(xf, n2g + i * 192, n2b + i * 192,
                                          h16, 0, 0);
    gemm_wmma<EPI_GELU_F16><<<gB(NTOK, 768), 256, 0, stream>>>(
        h16, 192, fc1T[i], 192, fc1_b + i * 768, NTOK, 768, 192,
        hid16, nullptr, nullptr, nullptr, 0);
    gemm_wmma<EPI_RESID_TOK><<<gB(NTOK, 192), 256, 0, stream>>>(
        hid16, 768, fc2T[i], 768, fc2_b + i * 192, NTOK, 192, 768,
        nullptr, xf, nullptr, nullptr, 0);
  }

  dwconv<<<(NTOK * CCH + 255) / 256, 256, 0, stream>>>(xf, dw_w, dw_b, h16);
  gemm_wmma<EPI_CONV><<<gB(NTOK, 192), 256, 0, stream>>>(
      h16, 192, pw16, 192, pw_b, NTOK, 192, 192,
      nullptr, (float*)d_out, bn_g, bn_b, 0);
}
